// ContentEncoder_72189810311839
// MI455X (gfx1250) — matrix operand order repaired and checked by reading, not compiled, not run on hardware
//
#include <hip/hip_runtime.h>

typedef __attribute__((ext_vector_type(16))) __bf16 v16bf;
typedef __attribute__((ext_vector_type(8)))  __bf16 v8bf;
typedef __attribute__((ext_vector_type(8)))  float  v8f;
typedef __attribute__((ext_vector_type(4)))  float  v4f;

// ---------------- fragment loaders (wave32, 16x16x32 bf16 WMMA) ----------------
// A (16x32, MxK): lane (r=lane&15, half=lane>>4) holds row M=r,
//   elements 0..7  = K[k0+8*half .. +7], elements 8..15 = K[k0+16+8*half .. +7]
__device__ __forceinline__ v16bf load_a_bf16(const __bf16* row, int k0, int half) {
  v8bf lo = *(const v8bf*)(row + k0 + 8 * half);
  v8bf hi = *(const v8bf*)(row + k0 + 16 + 8 * half);
  v16bf a;
#pragma unroll
  for (int i = 0; i < 8; ++i) { a[i] = lo[i]; a[i + 8] = hi[i]; }
  return a;
}

// B (32x16, KxN): lane holds column N=r; elements 0..7 = K[k0+8*half .. +7], elements 8..15 = K[k0+16+8*half .. +7]
__device__ __forceinline__ v16bf load_b_bf16(const __bf16* __restrict__ row, int k0, int half) {
  v8bf lo = *(const v8bf*)(row + k0 + 8 * half);
  v8bf hi = *(const v8bf*)(row + k0 + 16 + 8 * half);
  v16bf b;
#pragma unroll
  for (int i = 0; i < 8; ++i) { b[i] = lo[i]; b[i + 8] = hi[i]; }
  return b;
}

// ---------------- weight transpose + bf16 convert: W[K][512] -> Wt[512][K] ----------------
__global__ void transpose_w_kernel(const float* __restrict__ W, __bf16* __restrict__ Wt, int K) {
  long long idx = (long long)blockIdx.x * blockDim.x + threadIdx.x;
  long long total = 512LL * K;
  if (idx >= total) return;
  int n = (int)(idx / K);
  int k = (int)(idx % K);
  Wt[idx] = (__bf16)W[(size_t)k * 512 + n];
}

// stage 16 rows of an f32 [Nn x K] matrix into LDS as bf16 with padded stride (K+8)
__device__ __forceinline__ void stage16_f32_to_lds(__bf16* lA, const float* __restrict__ src,
                                                   int K, long long m0, long long Nn, int tid) {
  const int lda = K + 8;
  const int vecs = (16 * K) / 8;
  for (int i = tid; i < vecs; i += 256) {
    int flat = i * 8;
    int row = flat / K;
    int col = flat - row * K;
    long long ar = m0 + row;
    if (ar >= Nn) ar = Nn - 1;
    const v4f* p = (const v4f*)(src + ar * (long long)K + col);
    v4f x0 = p[0], x1 = p[1];
    v8bf o;
#pragma unroll
    for (int j = 0; j < 4; ++j) { o[j] = (__bf16)x0[j]; o[j + 4] = (__bf16)x1[j]; }
    *(v8bf*)(lA + (size_t)row * lda + col) = o;
  }
}

// one GEMM phase: acc[4] += lA(16xK) @ Bt'(cols n0..n0+63)
__device__ __forceinline__ void phase_gemm16(const __bf16* lA, const __bf16* __restrict__ Bt,
                                             int K, int n0, int r, int half, v8f acc[4]) {
  const int lda = K + 8;
  const __bf16* arow = lA + (size_t)r * lda;
  for (int k0 = 0; k0 < K; k0 += 32) {
    v16bf a = load_a_bf16(arow, k0, half);
#pragma unroll
    for (int t = 0; t < 4; ++t) {
      v16bf b = load_b_bf16(Bt + (size_t)(n0 + 16 * t + r) * K, k0, half);
      acc[t] = __builtin_amdgcn_wmma_f32_16x16x32_bf16(false, a, false, b, (short)0, acc[t],
                                                       false, false);
    }
  }
}

// ---------------- embed: h = max(E_order[o], E_tag[t], text@Wt+bt, img@Wi+bi, bg@Wb+bb) ----
__global__ void embed_kernel(const float* __restrict__ text, const float* __restrict__ img,
                             const float* __restrict__ bgimg,
                             const __bf16* __restrict__ wt_text, const __bf16* __restrict__ wt_img,
                             const __bf16* __restrict__ wt_bg,
                             const float* __restrict__ b_text, const float* __restrict__ b_img,
                             const float* __restrict__ b_bg,
                             const int* __restrict__ order, const int* __restrict__ tag,
                             const float* __restrict__ E_order, const float* __restrict__ E_tag,
                             float* __restrict__ h, long long Nn) {
  extern __shared__ __bf16 lA[];  // 16 * (768+8) bf16
  const int tid = threadIdx.x;
  const int lane = tid & 31;
  const int wave = tid >> 5;
  const int half = lane >> 4;
  const int r = lane & 15;
  const long long m0 = (long long)blockIdx.x * 16;
  const int n0 = wave * 64;

  float hm[4][8];
  v8f acc[4];

  // ---- text term (K = 768) ----
  stage16_f32_to_lds(lA, text, 768, m0, Nn, tid);
  __syncthreads();
#pragma unroll
  for (int t = 0; t < 4; ++t)
#pragma unroll
    for (int v = 0; v < 8; ++v) acc[t][v] = 0.0f;
  phase_gemm16(lA, wt_text, 768, n0, r, half, acc);
#pragma unroll
  for (int t = 0; t < 4; ++t) {
    float bs = b_text[n0 + 16 * t + r];
#pragma unroll
    for (int v = 0; v < 8; ++v) hm[t][v] = acc[t][v] + bs;
  }

  // ---- img term (K = 512) ----
  __syncthreads();
  stage16_f32_to_lds(lA, img, 512, m0, Nn, tid);
  __syncthreads();
#pragma unroll
  for (int t = 0; t < 4; ++t)
#pragma unroll
    for (int v = 0; v < 8; ++v) acc[t][v] = 0.0f;
  phase_gemm16(lA, wt_img, 512, n0, r, half, acc);
#pragma unroll
  for (int t = 0; t < 4; ++t) {
    float bs = b_img[n0 + 16 * t + r];
#pragma unroll
    for (int v = 0; v < 8; ++v) hm[t][v] = fmaxf(hm[t][v], acc[t][v] + bs);
  }

  // ---- bgimg term (K = 512) ----
  __syncthreads();
  stage16_f32_to_lds(lA, bgimg, 512, m0, Nn, tid);
  __syncthreads();
#pragma unroll
  for (int t = 0; t < 4; ++t)
#pragma unroll
    for (int v = 0; v < 8; ++v) acc[t][v] = 0.0f;
  phase_gemm16(lA, wt_bg, 512, n0, r, half, acc);
#pragma unroll
  for (int t = 0; t < 4; ++t) {
    float bs = b_bg[n0 + 16 * t + r];
#pragma unroll
    for (int v = 0; v < 8; ++v) hm[t][v] = fmaxf(hm[t][v], acc[t][v] + bs);
  }

  // ---- gathers + store (D layout: VGPR v -> M = 8*half+v, lane -> N = n0+16t+r) ----
#pragma unroll
  for (int t = 0; t < 4; ++t) {
    int n = n0 + 16 * t + r;
#pragma unroll
    for (int v = 0; v < 8; ++v) {
      long long m = m0 + 8 * half + v;
      if (m < Nn) {
        int o = order[m];
        int g = tag[m];
        float val = hm[t][v];
        val = fmaxf(val, E_order[(size_t)o * 512 + n]);
        val = fmaxf(val, E_tag[(size_t)g * 512 + n]);
        h[m * 512 + n] = val;
      }
    }
  }
}

// ---------------- MLP GEMM: out = A(bf16)[M x K] @ Bt' + bias ; optional relu --------------
// 32-row strip per block staged in LDS; 8 waves; each wave: 2 M-tiles x 4 N-tiles,
// B fragment shared across the 2 M-tiles. Bt is [512][K] bf16 (row n = column n of W).
__global__ void mlp_gemm_kernel(const __bf16* __restrict__ A, int M, int K,
                                const __bf16* __restrict__ Bt, const float* __restrict__ bias,
                                float* __restrict__ outF, __bf16* __restrict__ outB, int relu,
                                long long row_off) {
  extern __shared__ __bf16 lA[];  // 32 * (K+8) bf16
  const int tid = threadIdx.x;
  const int lane = tid & 31;
  const int wave = tid >> 5;
  const int half = lane >> 4;
  const int r = lane & 15;
  const long long m0 = (long long)blockIdx.x * 32;
  const int n0 = wave * 64;
  const int lda = K + 8;

  // cooperative stage of the 32 x K bf16 A strip
  {
    const int vecs = (32 * K) / 8;
    for (int i = tid; i < vecs; i += 256) {
      int flat = i * 8;
      int row = flat / K;
      int col = flat - row * K;
      long long ar = m0 + row;
      if (ar >= M) ar = M - 1;
      *(v8bf*)(lA + (size_t)row * lda + col) = *(const v8bf*)(A + ar * (long long)K + col);
    }
  }
  __syncthreads();

  v8f acc[2][4];
#pragma unroll
  for (int mt = 0; mt < 2; ++mt)
#pragma unroll
    for (int t = 0; t < 4; ++t)
#pragma unroll
      for (int v = 0; v < 8; ++v) acc[mt][t][v] = 0.0f;

  const __bf16* arow0 = lA + (size_t)r * lda;
  const __bf16* arow1 = lA + (size_t)(16 + r) * lda;
  for (int k0 = 0; k0 < K; k0 += 32) {
    v16bf a0 = load_a_bf16(arow0, k0, half);
    v16bf a1 = load_a_bf16(arow1, k0, half);
#pragma unroll
    for (int t = 0; t < 4; ++t) {
      v16bf b = load_b_bf16(Bt + (size_t)(n0 + 16 * t + r) * K, k0, half);
      acc[0][t] = __builtin_amdgcn_wmma_f32_16x16x32_bf16(false, a0, false, b, (short)0,
                                                          acc[0][t], false, false);
      acc[1][t] = __builtin_amdgcn_wmma_f32_16x16x32_bf16(false, a1, false, b, (short)0,
                                                          acc[1][t], false, false);
    }
  }

#pragma unroll
  for (int mt = 0; mt < 2; ++mt) {
#pragma unroll
    for (int t = 0; t < 4; ++t) {
      int n = n0 + 16 * t + r;
      float bs = bias[n];
#pragma unroll
      for (int v = 0; v < 8; ++v) {
        long long m = m0 + 16 * mt + 8 * half + v;
        if (m < M) {
          float val = acc[mt][t][v] + bs;
          if (relu) val = fmaxf(val, 0.0f);
          if (outB)
            outB[m * 512 + n] = (__bf16)val;
          else
            outF[(row_off + m) * 512 + n] = val;
        }
      }
    }
  }
}

// ---------------- init: u[leaf rows] = h_leaf ; d_out[0] = h_root ----------------
__global__ void init_kernel(float* __restrict__ u, float* __restrict__ dout,
                            const float* __restrict__ h_leaf, const float* __restrict__ h_root,
                            long long leaf0, long long nleaf) {
  long long idx = (long long)blockIdx.x * blockDim.x + threadIdx.x;
  long long total = nleaf * 512;
  if (idx < total) u[leaf0 * 512 + idx] = h_leaf[idx & 511];
  if (idx < 512) dout[idx] = h_root[idx];
}

// ---------------- prep upward x = [h[parent], max_children(u)] as bf16 ----------------
__global__ void prep_up_x_kernel(const float* __restrict__ h, const float* __restrict__ u,
                                 __bf16* __restrict__ xb, long long p0, long long c0,
                                 long long npar) {
  long long idx = (long long)blockIdx.x * blockDim.x + threadIdx.x;
  if (idx >= npar * 512) return;
  long long m = idx >> 9;
  int j = (int)(idx & 511);
  xb[m * 1024 + j] = (__bf16)h[(p0 + m) * 512 + j];
  const float* cb = u + (c0 + 4 * m) * 512 + j;
  float mx = cb[0];
  mx = fmaxf(mx, cb[512]);
  mx = fmaxf(mx, cb[1024]);
  mx = fmaxf(mx, cb[1536]);
  xb[m * 1024 + 512 + j] = (__bf16)mx;
}

// ---------------- prep downward x = [u[child], d[parent]] as bf16 ----------------
__global__ void prep_down_x_kernel(const float* __restrict__ u, const float* __restrict__ dout,
                                   __bf16* __restrict__ xb, long long c0, long long p0,
                                   long long nch) {
  long long idx = (long long)blockIdx.x * blockDim.x + threadIdx.x;
  if (idx >= nch * 512) return;
  long long m = idx >> 9;
  int j = (int)(idx & 511);
  xb[m * 1024 + j] = (__bf16)u[(c0 + m) * 512 + j];
  xb[m * 1024 + 512 + j] = (__bf16)dout[(p0 + (m >> 2)) * 512 + j];
}

// ---------------- residual: d_out += h ----------------
__global__ void final_add_kernel(float* __restrict__ dout, const float* __restrict__ h,
                                 long long total) {
  long long idx = (long long)blockIdx.x * blockDim.x + threadIdx.x;
  if (idx < total) dout[idx] += h[idx];
}

// =====================================================================================
extern "C" void kernel_launch(void* const* d_in, const int* in_sizes, int n_in, void* d_out,
                              int out_size, void* d_ws, size_t ws_size, hipStream_t stream) {
  (void)in_sizes; (void)n_in; (void)out_size; (void)ws_size;

  const int*   order  = (const int*)d_in[0];
  const int*   tag    = (const int*)d_in[1];
  const float* text   = (const float*)d_in[2];
  const float* img    = (const float*)d_in[3];
  const float* bgimg  = (const float*)d_in[4];
  const float* E_order = (const float*)d_in[5];
  const float* E_tag  = (const float*)d_in[6];
  const float* W_text = (const float*)d_in[7];
  const float* b_text = (const float*)d_in[8];
  const float* W_img  = (const float*)d_in[9];
  const float* b_img  = (const float*)d_in[10];
  const float* W_bg   = (const float*)d_in[11];
  const float* b_bg   = (const float*)d_in[12];
  const float* h_leaf = (const float*)d_in[13];
  const float* h_root = (const float*)d_in[14];
  const float* W1     = (const float*)d_in[15];
  const float* b1     = (const float*)d_in[16];
  const float* W2     = (const float*)d_in[17];
  const float* b2     = (const float*)d_in[18];
  float* dout = (float*)d_out;

  const long long Nn = 87381;  // (4^9 - 1) / 3

  // workspace carve-out (256B aligned regions)
  char* ws = (char*)d_ws;
  size_t off = 0;
  auto carve = [&](size_t bytes) -> char* {
    char* p = ws + off;
    off += (bytes + 255) & ~(size_t)255;
    return p;
  };
  float*  h_buf   = (float*)carve((size_t)Nn * 512 * 4);
  float*  u_buf   = (float*)carve((size_t)Nn * 512 * 4);
  __bf16* xb      = (__bf16*)carve((size_t)65536 * 1024 * 2);
  __bf16* y1      = (__bf16*)carve((size_t)65536 * 512 * 2);
  __bf16* wt_text = (__bf16*)carve((size_t)512 * 768 * 2);
  __bf16* wt_img  = (__bf16*)carve((size_t)512 * 512 * 2);
  __bf16* wt_bg   = (__bf16*)carve((size_t)512 * 512 * 2);
  __bf16* w1t     = (__bf16*)carve((size_t)512 * 1024 * 2);
  __bf16* w2t     = (__bf16*)carve((size_t)512 * 512 * 2);

  // 1) bf16 transposed weight caches
  transpose_w_kernel<<<(512 * 768 + 255) / 256, 256, 0, stream>>>(W_text, wt_text, 768);
  transpose_w_kernel<<<(512 * 512 + 255) / 256, 256, 0, stream>>>(W_img, wt_img, 512);
  transpose_w_kernel<<<(512 * 512 + 255) / 256, 256, 0, stream>>>(W_bg, wt_bg, 512);
  transpose_w_kernel<<<(512 * 1024 + 255) / 256, 256, 0, stream>>>(W1, w1t, 1024);
  transpose_w_kernel<<<(512 * 512 + 255) / 256, 256, 0, stream>>>(W2, w2t, 512);

  // 2) embed: h   (LDS: 16 rows x (768+8) bf16)
  {
    size_t sh = (size_t)16 * (768 + 8) * sizeof(__bf16);
    embed_kernel<<<(unsigned)((Nn + 15) / 16), 256, sh, stream>>>(
        text, img, bgimg, wt_text, wt_img, wt_bg, b_text, b_img, b_bg, order, tag, E_order,
        E_tag, h_buf, Nn);
  }

  // 3) init leaves (level 8: offset 21845, 65536 nodes) and root d
  {
    long long total = 65536LL * 512;
    init_kernel<<<(unsigned)((total + 255) / 256), 256, 0, stream>>>(u_buf, dout, h_leaf, h_root,
                                                                     21845, 65536);
  }

  auto level_off = [](int l) -> long long {
    long long r = 0, p = 1;
    for (int i = 0; i < l; ++i) { r += p; p *= 4; }
    return r;  // (4^l - 1) / 3
  };

  const size_t sh1 = (size_t)32 * (1024 + 8) * sizeof(__bf16);  // mlp1 LDS
  const size_t sh2 = (size_t)32 * (512 + 8) * sizeof(__bf16);   // mlp2 LDS

  // 4) upward pass: l = 7 .. 0
  for (int l = 7; l >= 0; --l) {
    long long p0 = level_off(l);
    long long npar = 1LL << (2 * l);
    long long c0 = level_off(l + 1);
    long long tot = npar * 512;
    prep_up_x_kernel<<<(unsigned)((tot + 255) / 256), 256, 0, stream>>>(h_buf, u_buf, xb, p0, c0,
                                                                        npar);
    mlp_gemm_kernel<<<(unsigned)((npar + 31) / 32), 256, sh1, stream>>>(
        xb, (int)npar, 1024, w1t, b1, nullptr, y1, 1, 0);
    mlp_gemm_kernel<<<(unsigned)((npar + 31) / 32), 256, sh2, stream>>>(
        y1, (int)npar, 512, w2t, b2, u_buf, nullptr, 0, p0);
  }

  // 5) downward pass: l = 1 .. 8 (writes directly into d_out rows)
  for (int l = 1; l <= 8; ++l) {
    long long c0 = level_off(l);
    long long nch = 1LL << (2 * l);
    long long p0 = level_off(l - 1);
    long long tot = nch * 512;
    prep_down_x_kernel<<<(unsigned)((tot + 255) / 256), 256, 0, stream>>>(u_buf, dout, xb, c0, p0,
                                                                          nch);
    mlp_gemm_kernel<<<(unsigned)((nch + 31) / 32), 256, sh1, stream>>>(
        xb, (int)nch, 1024, w1t, b1, nullptr, y1, 1, 0);
    mlp_gemm_kernel<<<(unsigned)((nch + 31) / 32), 256, sh2, stream>>>(
        y1, (int)nch, 512, w2t, b2, dout, nullptr, 0, c0);
  }

  // 6) residual
  {
    long long total = Nn * 512;
    final_add_kernel<<<(unsigned)((total + 255) / 256), 256, 0, stream>>>(dout, h_buf, total);
  }
}
